// LSTM_36936718745944
// MI455X (gfx1250) — matrix operand order repaired and checked by reading, not compiled, not run on hardware
//
#include <hip/hip_runtime.h>
#include <hip/hip_bf16.h>

// ---------------------------------------------------------------------------
// 2-layer LSTM for MI455X (gfx1250), bf16 WMMA 16x16x32 with f32 accumulate.
// T=512, B=64, D=256, H=512, L=2.
// Fused per-timestep kernel: [64x2048] gates GEMM + activations + c/h update.
// Software-pipelined (depth 2) K loop to keep ~20 b128 loads in flight.
// ---------------------------------------------------------------------------

typedef __attribute__((ext_vector_type(16))) __bf16 v16bf;
typedef __attribute__((ext_vector_type(8)))  float  v8f;
typedef __attribute__((ext_vector_type(4)))  unsigned int v4u;

#define T_STEPS 512
#define BATCH   64
#define DIN     256
#define HID     512

__device__ __forceinline__ float sigmoid_f(float x) {
  return 1.0f / (1.0f + __expf(-x));
}
__device__ __forceinline__ float tanh_f(float x) {
  // tanh(x) = 1 - 2/(exp(2x)+1); stable at both tails.
  return 1.0f - 2.0f / (__expf(2.0f * x) + 1.0f);
}

__device__ __forceinline__ v8f wmma_bf16(v16bf a, v16bf b, v8f c) {
  return __builtin_amdgcn_wmma_f32_16x16x32_bf16(
      /*neg_a=*/false, a, /*neg_b=*/false, b,
      /*c_mod=*/(short)0, c, /*reuse_a=*/false, /*reuse_b=*/false);
}

// A fragment (16x32 bf16, M x K), CDNA5 layout:
// lane l: m = l%16, hi = l/16; VGPR0-3 hold K = {0..7}+8*hi, VGPR4-7 hold K = {16..23}+8*hi.
// Per lane: two contiguous 16-byte chunks at k0+8*hi and k0+16+8*hi.
__device__ __forceinline__ v16bf load_A(const __bf16* __restrict__ X, int ld,
                                        int row, int k0, int hi) {
  union { v16bf v; v4u q[2]; } r;
  const __bf16* p = X + row * ld + k0 + hi * 8;
  r.q[0] = *(const v4u*)(p);
  r.q[1] = *(const v4u*)(p + 16);
  return r.v;
}

// B fragment from pre-packed weights: each lane's 16 bf16 are contiguous.
__device__ __forceinline__ v16bf load_B(const __bf16* __restrict__ p) {
  union { v16bf v; v4u q[2]; } r;
  r.q[0] = *(const v4u*)(p);
  r.q[1] = *(const v4u*)(p + 8);
  return r.v;
}

struct Frags {
  v16bf a, b0, b1, b2, b3;
};

// -------------------------- prep kernels -----------------------------------

__global__ __launch_bounds__(256) void cvt_bf16_kernel(__bf16* __restrict__ dst,
                                                       const float* __restrict__ src,
                                                       int n) {
  int i = blockIdx.x * blockDim.x + threadIdx.x;
  if (i < n) dst[i] = (__bf16)src[i];
}

__global__ __launch_bounds__(256) void bias_comb_kernel(float* __restrict__ dst,
                                                        const float* __restrict__ a,
                                                        const float* __restrict__ b,
                                                        int n) {
  int i = blockIdx.x * blockDim.x + threadIdx.x;
  if (i < n) dst[i] = a[i] + b[i];
}

// Pack W[N,K] (row-major fp32, W[n,k]) into bf16 B-fragment order:
// frag (nt,kt): dst[((nt*KT + kt)*512) + lane*16 + e] = W[nt*16 + (lane&15),
//    kt*32 + (e&7) + 8*(lane>>4) + 16*(e>>3)]
// (B 32x16 layout matches A: lane half hi holds K={0..7}+8*hi then K={16..23}+8*hi.)
__global__ __launch_bounds__(256) void pack_w_kernel(__bf16* __restrict__ dst,
                                                     const float* __restrict__ src,
                                                     int K, int total) {
  int o = blockIdx.x * blockDim.x + threadIdx.x;
  if (o >= total) return;
  const int KT = K >> 5;
  int e    = o & 15;
  int lane = (o >> 4) & 31;
  int f    = o >> 9;
  int kt   = f % KT;
  int nt   = f / KT;
  int nl = lane & 15, hi = lane >> 4;
  int kl = (e & 7) + 8 * hi + 16 * (e >> 3);
  dst[o] = (__bf16)src[(nt * 16 + nl) * K + kt * 32 + kl];
}

// -------------------------- fused LSTM step --------------------------------
// One wave (= one workgroup) computes a 16(batch) x 16(hidden) slice of all
// four gates, then applies activations and updates c/h in-register.
// 128 single-wave blocks cover B=64 x H=512 and spread across 128 WGPs.
template <int KXT>  // number of 32-wide K tiles in the input (x) GEMM
__global__ __launch_bounds__(32) void lstm_step_kernel(
    const __bf16* __restrict__ X,        // [B, KXT*32] row-major bf16
    const __bf16* __restrict__ Hprev,    // [B, 512]    row-major bf16
    const __bf16* __restrict__ Wihp,     // packed [128 ntiles][KXT ktiles][512]
    const __bf16* __restrict__ Whhp,     // packed [128 ntiles][16  ktiles][512]
    const float*  __restrict__ biasc,    // [2048] = b_ih + b_hh (i,f,c,o)
    float*        __restrict__ c_state,  // [B, 512] fp32 (in-place, exclusive)
    __bf16*       __restrict__ h_bf16_out, // [B, 512] bf16 (next step's A)
    float*        __restrict__ y_out,      // nullable: [B, 512] fp32
    float*        __restrict__ h_final,    // nullable
    float*        __restrict__ c_final) {  // nullable
  const int lane   = threadIdx.x;          // one wave per block
  const int wave   = blockIdx.x;           // 0..127
  const int m_tile = wave >> 5;            // 0..3  (batch tile)
  const int h_tile = wave & 31;            // 0..31 (hidden tile)
  const int nl = lane & 15, hi = lane >> 4;
  const int row = m_tile * 16 + nl;        // A-matrix row (batch index)

  constexpr int NT = KXT + 16;             // total K tiles (x part + h part)
  const int GI = 32 * KXT * 512;           // gate stride in packed W_ih
  const int GH = 32 * 16  * 512;           // gate stride in packed W_hh
  const __bf16* wihB = Wihp + (h_tile * KXT) * 512 + lane * 16;
  const __bf16* whhB = Whhp + (h_tile * 16)  * 512 + lane * 16;

  auto ld_tile = [&](int kt) -> Frags {
    Frags f;
    if (kt < KXT) {
      f.a = load_A(X, KXT * 32, row, kt * 32, hi);
      const __bf16* pb = wihB + kt * 512;
      f.b0 = load_B(pb);
      f.b1 = load_B(pb + GI);
      f.b2 = load_B(pb + 2 * GI);
      f.b3 = load_B(pb + 3 * GI);
    } else {
      const int k2 = kt - KXT;
      f.a = load_A(Hprev, HID, row, k2 * 32, hi);
      const __bf16* pb = whhB + k2 * 512;
      f.b0 = load_B(pb);
      f.b1 = load_B(pb + GH);
      f.b2 = load_B(pb + 2 * GH);
      f.b3 = load_B(pb + 3 * GH);
    }
    return f;
  };

  v8f acc0 = {}, acc1 = {}, acc2 = {}, acc3 = {};

  // depth-2 software pipeline: tiles kt and kt+1 resident, kt+2 loading
  Frags f0 = ld_tile(0);
  Frags f1 = ld_tile(1);
#pragma unroll
  for (int kt = 0; kt < NT; ++kt) {
    Frags f2;
    if (kt + 2 < NT) f2 = ld_tile(kt + 2);
    acc0 = wmma_bf16(f0.a, f0.b0, acc0);
    acc1 = wmma_bf16(f0.a, f0.b1, acc1);
    acc2 = wmma_bf16(f0.a, f0.b2, acc2);
    acc3 = wmma_bf16(f0.a, f0.b3, acc3);
    f0 = f1;
    if (kt + 2 < NT) f1 = f2;
  }

  // ---- activations + state update (C/D layout: lane n=l%16, VGPR v -> row v+8*hi)
  const int col = h_tile * 16 + nl;
  const float b_i = biasc[col];
  const float b_f = biasc[512 + col];
  const float b_c = biasc[1024 + col];
  const float b_o = biasc[1536 + col];

#pragma unroll
  for (int v = 0; v < 8; ++v) {
    const int rb  = m_tile * 16 + v + 8 * hi;  // batch index
    const int idx = rb * HID + col;
    float ig = sigmoid_f(acc0[v] + b_i);
    float fg = sigmoid_f(acc1[v] + b_f);
    float cg = tanh_f(acc2[v] + b_c);
    float og = sigmoid_f(acc3[v] + b_o);
    float cp = c_state[idx];
    float cn = fg * cp + ig * cg;
    float hn = og * tanh_f(cn);
    c_state[idx]    = cn;
    h_bf16_out[idx] = (__bf16)hn;
    if (y_out)   y_out[idx]   = hn;
    if (h_final) h_final[idx] = hn;
    if (c_final) c_final[idx] = cn;
  }
}

// -------------------------- host driver ------------------------------------

extern "C" void kernel_launch(void* const* d_in, const int* in_sizes, int n_in,
                              void* d_out, int out_size, void* d_ws, size_t ws_size,
                              hipStream_t stream) {
  (void)in_sizes; (void)n_in; (void)out_size; (void)ws_size;
  const float* x     = (const float*)d_in[0];
  const float* h0    = (const float*)d_in[1];
  const float* c0    = (const float*)d_in[2];
  const float* w_ih0 = (const float*)d_in[3];
  const float* b_ih0 = (const float*)d_in[4];
  const float* w_hh0 = (const float*)d_in[5];
  const float* b_hh0 = (const float*)d_in[6];
  const float* w_ih1 = (const float*)d_in[7];
  const float* b_ih1 = (const float*)d_in[8];
  const float* w_hh1 = (const float*)d_in[9];
  const float* b_hh1 = (const float*)d_in[10];
  float* out = (float*)d_out;

  const size_t BH  = (size_t)BATCH * HID;           // 32768
  const size_t YSZ = (size_t)T_STEPS * BH;          // 16,777,216

  // workspace carve-up (256B aligned)
  char* wsp = (char*)d_ws;
  auto carve = [&](size_t bytes) -> void* {
    void* p = (void*)wsp;
    wsp += (bytes + 255) & ~(size_t)255;
    return p;
  };
  __bf16* x_b   = (__bf16*)carve((size_t)T_STEPS * BATCH * DIN * 2);
  __bf16* y0_b  = (__bf16*)carve((size_t)T_STEPS * BH * 2);
  __bf16* wih0p = (__bf16*)carve((size_t)4 * HID * DIN * 2);
  __bf16* whh0p = (__bf16*)carve((size_t)4 * HID * HID * 2);
  __bf16* wih1p = (__bf16*)carve((size_t)4 * HID * HID * 2);
  __bf16* whh1p = (__bf16*)carve((size_t)4 * HID * HID * 2);
  float*  bias0 = (float*)carve((size_t)4 * HID * 4);
  float*  bias1 = (float*)carve((size_t)4 * HID * 4);
  float*  cs0   = (float*)carve(BH * 4);
  float*  cs1   = (float*)carve(BH * 4);
  __bf16* h0b   = (__bf16*)carve(BH * 2);
  __bf16* h1b0  = (__bf16*)carve(BH * 2);
  __bf16* h1b1  = (__bf16*)carve(BH * 2);

  // ---- prep: conversions, weight packing, bias fusion, state init ----
  {
    int n = T_STEPS * BATCH * DIN;
    cvt_bf16_kernel<<<(n + 255) / 256, 256, 0, stream>>>(x_b, x, n);
  }
  cvt_bf16_kernel<<<((int)BH + 255) / 256, 256, 0, stream>>>(h0b,  h0,      (int)BH);
  cvt_bf16_kernel<<<((int)BH + 255) / 256, 256, 0, stream>>>(h1b0, h0 + BH, (int)BH);
  {
    int n0 = 4 * HID * DIN;   // w_ih0
    int n1 = 4 * HID * HID;   // others
    pack_w_kernel<<<(n0 + 255) / 256, 256, 0, stream>>>(wih0p, w_ih0, DIN, n0);
    pack_w_kernel<<<(n1 + 255) / 256, 256, 0, stream>>>(whh0p, w_hh0, HID, n1);
    pack_w_kernel<<<(n1 + 255) / 256, 256, 0, stream>>>(wih1p, w_ih1, HID, n1);
    pack_w_kernel<<<(n1 + 255) / 256, 256, 0, stream>>>(whh1p, w_hh1, HID, n1);
  }
  bias_comb_kernel<<<(4 * HID + 255) / 256, 256, 0, stream>>>(bias0, b_ih0, b_hh0, 4 * HID);
  bias_comb_kernel<<<(4 * HID + 255) / 256, 256, 0, stream>>>(bias1, b_ih1, b_hh1, 4 * HID);
  (void)hipMemcpyAsync(cs0, c0,      BH * 4, hipMemcpyDeviceToDevice, stream);
  (void)hipMemcpyAsync(cs1, c0 + BH, BH * 4, hipMemcpyDeviceToDevice, stream);

  // d_out layout: y1 [T,B,H] | h_out [2,B,H] | c_out [2,B,H]
  float* hout0 = out + YSZ;
  float* hout1 = out + YSZ + BH;
  float* cout0 = out + YSZ + 2 * BH;
  float* cout1 = out + YSZ + 3 * BH;

  // ---- layer 0: x (K=256) + h (K=512); h stream doubles as layer-1 input ----
  for (int t = 0; t < T_STEPS; ++t) {
    const __bf16* Xp = x_b + (size_t)t * BATCH * DIN;
    const __bf16* Hp = (t == 0) ? h0b : (y0_b + (size_t)(t - 1) * BH);
    __bf16* Hout = y0_b + (size_t)t * BH;
    const bool last = (t == T_STEPS - 1);
    lstm_step_kernel<DIN / 32><<<128, 32, 0, stream>>>(
        Xp, Hp, wih0p, whh0p, bias0, cs0, Hout,
        nullptr, last ? hout0 : nullptr, last ? cout0 : nullptr);
  }

  // ---- layer 1: x=y0 (K=512) + h (K=512); h ping-pongs; y1 -> d_out ----
  for (int t = 0; t < T_STEPS; ++t) {
    const __bf16* Xp = y0_b + (size_t)t * BH;
    const __bf16* Hp = (t & 1) ? h1b1 : h1b0;
    __bf16* Hout     = (t & 1) ? h1b0 : h1b1;
    const bool last = (t == T_STEPS - 1);
    lstm_step_kernel<HID / 32><<<128, 32, 0, stream>>>(
        Xp, Hp, wih1p, whh1p, bias1, cs1, Hout,
        out + (size_t)t * BH, last ? hout1 : nullptr, last ? cout1 : nullptr);
  }
}
